// ChannelAttentionSequence_53566832115937
// MI455X (gfx1250) — matrix operand order repaired and checked by reading, not compiled, not run on hardware
//
#include <hip/hip_runtime.h>
#include <hip/hip_bf16.h>

// ---------------------------------------------------------------------------
// ChannelAttentionSequence on MI455X (gfx1250, wave32, WMMA f16->f32).
// Memory-bound problem (~77 GFLOP vs <1GB traffic @ 23.3TB/s), so:
//  * all GEMMs use v_wmma_f32_16x16x32_f16 (fp32 accumulate)
//  * one-shot fp32->f16 convert pass for activations + weights so the GEMM
//    inner loops are pure b128 loads + WMMA (no per-k-step v_cvt), and
//    activation re-reads happen at f16 width
//  * intermediates (Q,K,V,head-out) stored f16
//  * layouts chosen so every WMMA operand load is lane-contiguous
// ---------------------------------------------------------------------------

typedef __attribute__((ext_vector_type(16))) _Float16 v16h;
typedef __attribute__((ext_vector_type(8)))  _Float16 v8h;
typedef __attribute__((ext_vector_type(8)))  float    v8f;

#define B_   4
#define N_   8192
#define C_   512
#define H_   8
#define D_   64
#define BH_  32            // B*H
#define M_   32768         // B*N
#define NCHUNK 32          // split-K chunks over the sequence axis
#define CHUNK_N (N_ / NCHUNK)   // 256

static __device__ __forceinline__ v8f wmma_f16(v16h a, v16h b, v8f c) {
  // 8 args: (neg_a, A, neg_b, B, c_mod, C, reuse_a, reuse_b)
  return __builtin_amdgcn_wmma_f32_16x16x32_f16(false, a, false, b,
                                                (short)0, c, false, false);
}

// Per-lane WMMA operand k order used by this file (A and B share it, so it is valid):
//  A (16x32 f16): lane L holds row M=L%16; lanes 0-15 hold K=kb..kb+15,
//    lanes 16-31 hold K=kb+16..kb+31 (contiguous 16, same as B).
//  B (32x16 f16): lane L holds col N=L%16; lanes 0-15 hold K=kb..kb+15,
//    lanes 16-31 hold K=kb+16..kb+31 (contiguous 16).
//  C/D (16x16 f32): lane L holds col N=L%16; VGPR r holds row M=r (+8 for
//    lanes 16-31).

// ---------------------------------------------------------------------------
// Kernel 0: bulk fp32 -> f16 conversion (8 elements per thread).
// ---------------------------------------------------------------------------
__global__ __launch_bounds__(256) void cvt_f16_kernel(
    const float* __restrict__ src, _Float16* __restrict__ dst, int n8) {
  const int i = blockIdx.x * blockDim.x + threadIdx.x;
  if (i >= n8) return;
  const size_t base = (size_t)i * 8;
  v8h o;
#pragma unroll
  for (int k = 0; k < 8; ++k) o[k] = (_Float16)src[base + k];
  *(v8h*)(dst + base) = o;
}

// ---------------------------------------------------------------------------
// Kernel 1: projection GEMM, all-f16 operands, f16 output.
//   Y[m,c] = sum_k X[m,k]*W[c,k] + bias[c]
//   MODE 0 (Q,K): dst[((b*H+h)*D + d)*N + n] = Y*scale   (head-transposed)
//   MODE 1 (V):   dst[((b*H+h)*N + n)*D + d] = Y         (head-natural)
// One wave computes a 16-row x 128-col strip (8 accumulators); X re-read 4x
// at f16 width, f16 weights (0.5MB) live in L2.
// ---------------------------------------------------------------------------
template <int MODE>
__global__ __launch_bounds__(256) void proj_f16_kernel(
    const _Float16* __restrict__ X, const _Float16* __restrict__ W,
    const float* __restrict__ bias, _Float16* __restrict__ dst, float scale) {
  const int wave  = (blockIdx.x * blockDim.x + threadIdx.x) >> 5;
  const int lane  = threadIdx.x & 31;
  const int lr    = lane & 15;
  const int hf    = lane >> 4;
  const int strip = wave & 3;       // 4 strips of 128 cols
  const int mt    = wave >> 2;      // 2048 row tiles
  const int m0    = mt << 4;
  const int c0    = strip << 7;

  v8f acc[8];
#pragma unroll
  for (int j = 0; j < 8; ++j) acc[j] = {};

  const _Float16* xrow = X + (size_t)(m0 + lr) * C_;
  for (int kb = 0; kb < C_; kb += 32) {
    v16h a;
    const _Float16* xp = xrow + kb + hf * 16;
#pragma unroll
    for (int i = 0; i < 16; ++i) a[i] = xp[i];
#pragma unroll
    for (int j = 0; j < 8; ++j) {
      const int c = c0 + (j << 4) + lr;
      const _Float16* wp = W + (size_t)c * C_ + kb + hf * 16;
      v16h b;
#pragma unroll
      for (int i = 0; i < 16; ++i) b[i] = wp[i];
      acc[j] = wmma_f16(a, b, acc[j]);
    }
  }

  const int bb = m0 >> 13;                    // batch (N = 8192)
  const int n0 = (m0 & (N_ - 1)) + hf * 8;    // sequence base for this half
#pragma unroll
  for (int j = 0; j < 8; ++j) {
    const int c = c0 + (j << 4) + lr;
    const int h = c >> 6;
    const int d = c & 63;
    const float bv = bias[c];
    if (MODE == 0) {
      v8h st;
#pragma unroll
      for (int r = 0; r < 8; ++r) st[r] = (_Float16)((acc[j][r] + bv) * scale);
      *(v8h*)(dst + ((size_t)((bb * H_ + h) * D_ + d)) * N_ + n0) = st;   // 16B store
    } else {
#pragma unroll
      for (int r = 0; r < 8; ++r)
        dst[((size_t)(bb * H_ + h) * N_ + n0 + r) * D_ + d] = (_Float16)(acc[j][r] + bv);
    }
  }
}

// ---------------------------------------------------------------------------
// Kernel 2: attention logits, split-K over the sequence axis.
//   attP[bh][ch][d][e] = sum_{n in chunk} Qt[bh,d,n] * Kt[bh,e,n]
// (1/sqrt(D) already folded into Qt.)  16 waves per block = one (bh,chunk);
// each wave owns one 16x16 (d,e) tile, 8 WMMA k-steps of 32.
// ---------------------------------------------------------------------------
__global__ __launch_bounds__(512) void att_partial_kernel(
    const _Float16* __restrict__ Qt, const _Float16* __restrict__ Kt,
    float* __restrict__ attP) {
  const int bh = blockIdx.x >> 5;
  const int ch = blockIdx.x & 31;
  const int wv = threadIdx.x >> 5;      // 16 waves
  const int lane = threadIdx.x & 31;
  const int lr = lane & 15, hf = lane >> 4;
  const int dt = wv >> 2, et = wv & 3;

  const _Float16* qrow = Qt + ((size_t)bh * D_ + dt * 16 + lr) * N_;
  const _Float16* krow = Kt + ((size_t)bh * D_ + et * 16 + lr) * N_;

  v8f acc = {};
  const int nb = ch * CHUNK_N;
  for (int kb = nb; kb < nb + CHUNK_N; kb += 32) {
    v16h a, b;
    const _Float16* qp = qrow + kb + hf * 16;
#pragma unroll
    for (int i = 0; i < 16; ++i) a[i] = qp[i];
    const _Float16* kp = krow + kb + hf * 16;
#pragma unroll
    for (int i = 0; i < 16; ++i) b[i] = kp[i];
    acc = wmma_f16(a, b, acc);
  }

  float* outp = attP + (size_t)(bh * NCHUNK + ch) * (D_ * D_);
#pragma unroll
  for (int r = 0; r < 8; ++r)
    outp[(dt * 16 + hf * 8 + r) * D_ + et * 16 + lr] = acc[r];
}

// ---------------------------------------------------------------------------
// Kernel 3: reduce split-K partials + softmax over e (row length 64).
// One wave per (bh,d) row; 2 elements per lane; wave reductions via shfl_xor.
// Writes fp32 att into d_out and an f16 copy for the next GEMM.
// ---------------------------------------------------------------------------
__global__ __launch_bounds__(256) void softmax_kernel(
    const float* __restrict__ attP, float* __restrict__ attOut,
    _Float16* __restrict__ att16) {
  const int wave = (blockIdx.x * blockDim.x + threadIdx.x) >> 5;  // 2048 rows
  const int lane = threadIdx.x & 31;
  const int bh = wave >> 6, d = wave & 63;

  float v0 = 0.f, v1 = 0.f;
  for (int ch = 0; ch < NCHUNK; ++ch) {
    const float* p = attP + (size_t)(bh * NCHUNK + ch) * (D_ * D_) + d * D_;
    v0 += p[lane];
    v1 += p[lane + 32];
  }
  float mx = fmaxf(v0, v1);
#pragma unroll
  for (int off = 16; off >= 1; off >>= 1) mx = fmaxf(mx, __shfl_xor(mx, off, 32));
  float e0 = __expf(v0 - mx), e1 = __expf(v1 - mx);
  float s = e0 + e1;
#pragma unroll
  for (int off = 16; off >= 1; off >>= 1) s += __shfl_xor(s, off, 32);
  const float inv = 1.f / s;
  e0 *= inv; e1 *= inv;

  const size_t o = (size_t)bh * (D_ * D_) + (size_t)d * D_;
  attOut[o + lane] = e0;
  attOut[o + lane + 32] = e1;
  att16[o + lane] = (_Float16)e0;
  att16[o + lane + 32] = (_Float16)e1;
}

// ---------------------------------------------------------------------------
// Kernel 4: head output  O[b,n,h*64+d] = sum_e att16[bh,d,e] * V[bh,n,e]
// One wave per (bh, 16-col n-tile); V tile loaded once (B operand for both
// k-steps), 4 d-tiles x 2 WMMA each; 16B f16 store per lane.
// ---------------------------------------------------------------------------
__global__ __launch_bounds__(256) void av_kernel(
    const _Float16* __restrict__ att16, const _Float16* __restrict__ Vn,
    _Float16* __restrict__ O) {
  const int wave = (blockIdx.x * blockDim.x + threadIdx.x) >> 5;  // 16384
  const int lane = threadIdx.x & 31;
  const int lr = lane & 15, hf = lane >> 4;
  const int bh = wave >> 9;
  const int nt = wave & 511;
  const int n = nt * 16 + lr;

  const _Float16* vrow = Vn + ((size_t)bh * N_ + n) * D_;
  v16h b0, b1;
#pragma unroll
  for (int i = 0; i < 16; ++i) { b0[i] = vrow[hf * 16 + i]; b1[i] = vrow[32 + hf * 16 + i]; }

  const _Float16* am = att16 + (size_t)bh * (D_ * D_);
  const int bb = bh >> 3, h = bh & 7;
#pragma unroll
  for (int dt = 0; dt < 4; ++dt) {
    const _Float16* arow = am + (dt * 16 + lr) * D_;
    v16h a0, a1;
#pragma unroll
    for (int i = 0; i < 16; ++i) {
      a0[i] = arow[hf * 16 + i];
      a1[i] = arow[32 + hf * 16 + i];
    }
    v8f acc = {};
    acc = wmma_f16(a0, b0, acc);
    acc = wmma_f16(a1, b1, acc);
    v8h st;
#pragma unroll
    for (int r = 0; r < 8; ++r) st[r] = (_Float16)acc[r];
    *(v8h*)(O + ((size_t)(bb * N_ + n)) * C_ + h * D_ + dt * 16 + hf * 8) = st;
  }
}

// ---------------------------------------------------------------------------
// Kernel 5: final projection (f16 operands), fp32 output into d_out.
//   out[m,c] = sum_k O[m,k]*Wp[c,k] + bp[c]
// ---------------------------------------------------------------------------
__global__ __launch_bounds__(256) void proj_out_kernel(
    const _Float16* __restrict__ O, const _Float16* __restrict__ W,
    const float* __restrict__ bias, float* __restrict__ out) {
  const int wave  = (blockIdx.x * blockDim.x + threadIdx.x) >> 5;
  const int lane  = threadIdx.x & 31;
  const int lr    = lane & 15;
  const int hf    = lane >> 4;
  const int strip = wave & 3;
  const int mt    = wave >> 2;
  const int m0    = mt << 4;
  const int c0    = strip << 7;

  v8f acc[8];
#pragma unroll
  for (int j = 0; j < 8; ++j) acc[j] = {};

  const _Float16* orow = O + (size_t)(m0 + lr) * C_;
  for (int kb = 0; kb < C_; kb += 32) {
    v16h a;
    const _Float16* op = orow + kb + hf * 16;
#pragma unroll
    for (int i = 0; i < 16; ++i) a[i] = op[i];
#pragma unroll
    for (int j = 0; j < 8; ++j) {
      const int c = c0 + (j << 4) + lr;
      const _Float16* wp = W + (size_t)c * C_ + kb + hf * 16;
      v16h b;
#pragma unroll
      for (int i = 0; i < 16; ++i) b[i] = wp[i];
      acc[j] = wmma_f16(a, b, acc[j]);
    }
  }

#pragma unroll
  for (int j = 0; j < 8; ++j) {
    const int c = c0 + (j << 4) + lr;
    const float bv = bias[c];
#pragma unroll
    for (int r = 0; r < 8; ++r)
      out[(size_t)(m0 + hf * 8 + r) * C_ + c] = acc[j][r] + bv;
  }
}

// ---------------------------------------------------------------------------
// Workspace layout (d_ws), ~211 MB total:
//   [  0MB) Qt    [BH][D][N] f16   32MB   (scaled by 1/sqrt(D))
//   [ 32MB) Kt    [BH][D][N] f16   32MB
//   [ 64MB) Vn    [BH][N][D] f16   32MB
//   [ 96MB) O     [B*N][C]   f16   32MB
//   [128MB) attP  [BH][32][D][D] f32 16MB
//   [144MB) att16 [BH][D][D] f16  256KB (1MB slot)
//   [145MB) Xh    [B*N][C]   f16   32MB
//   [177MB) Ch    [B*N][C]   f16   32MB
//   [209MB) Wq16/Wk16/Wv16/Wp16  f16  4 x 512KB
// ---------------------------------------------------------------------------
extern "C" void kernel_launch(void* const* d_in, const int* in_sizes, int n_in,
                              void* d_out, int out_size, void* d_ws, size_t ws_size,
                              hipStream_t stream) {
  (void)in_sizes; (void)n_in; (void)out_size; (void)ws_size;

  const float* x       = (const float*)d_in[0];
  const float* context = (const float*)d_in[1];
  const float* Wq_w    = (const float*)d_in[2];
  const float* Wq_b    = (const float*)d_in[3];
  const float* Wk_w    = (const float*)d_in[4];
  const float* Wk_b    = (const float*)d_in[5];
  const float* Wv_w    = (const float*)d_in[6];
  const float* Wv_b    = (const float*)d_in[7];
  const float* proj_w  = (const float*)d_in[8];
  const float* proj_b  = (const float*)d_in[9];

  float* out    = (float*)d_out;                    // [B,N,C]
  float* attOut = out + (size_t)B_ * N_ * C_;       // [B,H,D,D]

  const size_t MB = (size_t)1024 * 1024;
  char* ws = (char*)d_ws;
  _Float16* Qt    = (_Float16*)(ws);
  _Float16* Kt    = (_Float16*)(ws + 32 * MB);
  _Float16* Vn    = (_Float16*)(ws + 64 * MB);
  _Float16* O     = (_Float16*)(ws + 96 * MB);
  float*    attP  = (float*)   (ws + 128 * MB);
  _Float16* att16 = (_Float16*)(ws + 144 * MB);
  _Float16* Xh    = (_Float16*)(ws + 145 * MB);
  _Float16* Ch    = (_Float16*)(ws + 177 * MB);
  _Float16* Wq16  = (_Float16*)(ws + 209 * MB);
  _Float16* Wk16  = (_Float16*)(ws + 209 * MB + 512 * 1024);
  _Float16* Wv16  = (_Float16*)(ws + 210 * MB);
  _Float16* Wp16  = (_Float16*)(ws + 210 * MB + 512 * 1024);

  const float qscale = 0.125f;  // 1/sqrt(D)
  dim3 blk(256);

  // fp32 -> f16 pre-pass (activations + weights)
  const int actN8 = (B_ * N_ * C_) / 8;   // 2,097,152
  const int wN8   = (C_ * C_) / 8;        // 32,768
  cvt_f16_kernel<<<actN8 / 256, blk, 0, stream>>>(x,       Xh,   actN8);
  cvt_f16_kernel<<<actN8 / 256, blk, 0, stream>>>(context, Ch,   actN8);
  cvt_f16_kernel<<<wN8 / 256,   blk, 0, stream>>>(Wq_w,    Wq16, wN8);
  cvt_f16_kernel<<<wN8 / 256,   blk, 0, stream>>>(Wk_w,    Wk16, wN8);
  cvt_f16_kernel<<<wN8 / 256,   blk, 0, stream>>>(Wv_w,    Wv16, wN8);
  cvt_f16_kernel<<<wN8 / 256,   blk, 0, stream>>>(proj_w,  Wp16, wN8);

  proj_f16_kernel<0><<<1024, blk, 0, stream>>>(Xh, Wq16, Wq_b, Qt, qscale);
  proj_f16_kernel<0><<<1024, blk, 0, stream>>>(Ch, Wk16, Wk_b, Kt, 1.0f);
  proj_f16_kernel<1><<<1024, blk, 0, stream>>>(Ch, Wv16, Wv_b, Vn, 1.0f);
  att_partial_kernel<<<1024, dim3(512), 0, stream>>>(Qt, Kt, attP);
  softmax_kernel<<<256, blk, 0, stream>>>(attP, attOut, att16);
  av_kernel<<<2048, blk, 0, stream>>>(att16, Vn, O);
  proj_out_kernel<<<1024, blk, 0, stream>>>(O, Wp16, proj_b, out);
}
